// Head_34668976014011
// MI455X (gfx1250) — compile-verified, matrix operand order checked by reading, not run on hardware
//
#include <hip/hip_runtime.h>
#include <hip/hip_bf16.h>

typedef __attribute__((ext_vector_type(16))) __bf16 v16bf;
typedef __attribute__((ext_vector_type(8)))  __bf16 v8bf;
typedef __attribute__((ext_vector_type(8)))  float  v8f;

namespace {
constexpr int kB = 8;
constexpr int kT = 2048;
constexpr int kC = 1024;
constexpr int kH = 64;
constexpr float kScale = 0.03125f;  // C^-0.5 = 1/32 (reference scales by full C)
}

// A/B fragment for v_wmma_f32_16x16x32_bf16 (wave32):
// lane holds row/col (lane&15); elements 0..7  = K(8*half .. 8*half+7),
//                               elements 8..15 = K(16+8*half .. 16+8*half+7).
// Both runs are contiguous in memory -> two 16-byte loads.
__device__ __forceinline__ v16bf frag_from_bf16(const __bf16* p, int half) {
  v8bf lo = *(const v8bf*)(p + 8 * half);
  v8bf hi = *(const v8bf*)(p + 16 + 8 * half);
  return __builtin_shufflevector(lo, hi, 0, 1, 2, 3, 4, 5, 6, 7,
                                 8, 9, 10, 11, 12, 13, 14, 15);
}

// Same fragment but sourced from fp32 memory with in-register bf16 convert.
__device__ __forceinline__ v16bf frag_from_f32(const float* p, int half) {
  v16bf r;
#pragma unroll
  for (int i = 0; i < 8; ++i) {
    r[i]     = (__bf16)p[8 * half + i];
    r[i + 8] = (__bf16)p[16 + 8 * half + i];
  }
  return r;
}

__device__ __forceinline__ v8f wmma_bf16(v16bf a, v16bf b, v8f c) {
  // (neg_a, A, neg_b, B, c_mod, C, reuse_a, reuse_b)
  return __builtin_amdgcn_wmma_f32_16x16x32_bf16(false, a, false, b,
                                                 (short)0, c, false, false);
}

// One-shot fp32 -> bf16 conversion of the three weight matrices into
// wb[3][H][C] so the GEMM hot loop does pure bf16 b128 loads (no cvt).
__global__ void cvt_w_kernel(const float* __restrict__ Wq,
                             const float* __restrict__ Wk,
                             const float* __restrict__ Wv,
                             __bf16* __restrict__ wb) {
  const int n = kH * kC;  // 65536 per matrix
  const int idx4 = (blockIdx.x * blockDim.x + threadIdx.x) * 4;
  if (idx4 >= 3 * n) return;
  const int wsel = idx4 / n;
  const int off  = idx4 - wsel * n;
  const float* src = (wsel == 0) ? Wq : (wsel == 1) ? Wk : Wv;
  const float4 f = *(const float4*)(src + off);
  __bf16* d = wb + (size_t)wsel * n + off;
  d[0] = (__bf16)f.x; d[1] = (__bf16)f.y;
  d[2] = (__bf16)f.z; d[3] = (__bf16)f.w;
}

// One wave computes the 16x64 q, k AND v tiles for 16 token rows, sharing a
// single x A-fragment across 12 WMMAs per K-step (x read once, not 3x).
// fp32 WMMA accumulation; q pre-scaled; v stored transposed [H,T] so the
// later P@V matmul's B-operand K-dimension is contiguous.
__global__ void qkv_proj_kernel(const float* __restrict__ x,
                                const __bf16* __restrict__ wb,   // [3][H][C]
                                __bf16* __restrict__ qb,
                                __bf16* __restrict__ kb,
                                __bf16* __restrict__ vtb) {
  const int lane = threadIdx.x & 31;
  const int row  = lane & 15;
  const int half = lane >> 4;
  const int tile = blockIdx.x;                 // B*T/16 tiles
  const int b  = tile / (kT / 16);
  const int t0 = (tile % (kT / 16)) * 16;

  const float* xrow = x + (size_t)(b * kT + t0 + row) * kC;

  v8f acc[3][4] = {};
  for (int k0 = 0; k0 < kC; k0 += 32) {
    const v16bf a = frag_from_f32(xrow + k0, half);
#pragma unroll
    for (int wsel = 0; wsel < 3; ++wsel) {
#pragma unroll
      for (int nt = 0; nt < 4; ++nt) {
        const __bf16* wrow =
            wb + ((size_t)wsel * kH + nt * 16 + row) * kC + k0;
        acc[wsel][nt] = wmma_bf16(a, frag_from_bf16(wrow, half), acc[wsel][nt]);
      }
    }
  }

#pragma unroll
  for (int wsel = 0; wsel < 3; ++wsel) {
    const float scl = (wsel == 0) ? kScale : 1.0f;
#pragma unroll
    for (int nt = 0; nt < 4; ++nt) {
#pragma unroll
      for (int r = 0; r < 8; ++r) {
        const int t = t0 + r + 8 * half;        // D layout: M = r + 8*half
        const int h = nt * 16 + row;            //           N = lane&15
        const __bf16 v = (__bf16)(acc[wsel][nt][r] * scl);
        if (wsel == 0)      qb[(size_t)(b * kT + t) * kH + h] = v;
        else if (wsel == 1) kb[(size_t)(b * kT + t) * kH + h] = v;
        else                vtb[((size_t)b * kH + h) * kT + t] = v;
      }
    }
  }
}

// Flash-attention: one wave per 16 query rows, 32-key blocks up to the diagonal.
__global__ void attn_kernel(const __bf16* __restrict__ qb,
                            const __bf16* __restrict__ kb,
                            const __bf16* __restrict__ vtb,
                            float* __restrict__ out) {
  __shared__ __bf16 plds[16 * 32] __attribute__((aligned(16)));
  const int lane = threadIdx.x & 31;
  const int row  = lane & 15;
  const int half = lane >> 4;
  const int b  = blockIdx.y;
  const int t0 = blockIdx.x * 16;

  const __bf16* qrow = qb + (size_t)(b * kT + t0 + row) * kH;
  const v16bf qa0 = frag_from_bf16(qrow, half);        // K = 0..31 of H
  const v16bf qa1 = frag_from_bf16(qrow + 32, half);   // K = 32..63 of H

  float mi[8], li[8];
#pragma unroll
  for (int r = 0; r < 8; ++r) { mi[r] = -3.0e38f; li[r] = 0.0f; }
  v8f o[4] = {};

  const int nblocks = (t0 + 16 + 31) / 32;             // key blocks to diagonal
  for (int jb = 0; jb < nblocks; ++jb) {
    const int j0 = jb * 32;

    // scores: 16 queries x 32 keys, K-dim = H = 64 (two 32-chunks)
    v8f s0v = {}, s1v = {};
    {
      const __bf16* krow0 = kb + (size_t)(b * kT + j0 + row) * kH;
      s0v = wmma_bf16(qa0, frag_from_bf16(krow0, half), s0v);
      s0v = wmma_bf16(qa1, frag_from_bf16(krow0 + 32, half), s0v);
      const __bf16* krow1 = kb + (size_t)(b * kT + j0 + 16 + row) * kH;
      s1v = wmma_bf16(qa0, frag_from_bf16(krow1, half), s1v);
      s1v = wmma_bf16(qa1, frag_from_bf16(krow1 + 32, half), s1v);
    }

    // causal mask + online softmax (row = M = r + 8*half, key col = lane&15)
    float alpha[8];
#pragma unroll
    for (int r = 0; r < 8; ++r) {
      const int m = t0 + r + 8 * half;
      float s0 = (j0 + row      <= m) ? s0v[r] : -1.0e30f;
      float s1 = (j0 + 16 + row <= m) ? s1v[r] : -1.0e30f;
      float rmax = fmaxf(s0, s1);
#pragma unroll
      for (int d = 1; d < 16; d <<= 1)          // reduce within 16-lane half
        rmax = fmaxf(rmax, __shfl_xor(rmax, d, 32));
      const float mnew = fmaxf(mi[r], rmax);
      const float p0 = __expf(s0 - mnew);
      const float p1 = __expf(s1 - mnew);
      float rsum = p0 + p1;
#pragma unroll
      for (int d = 1; d < 16; d <<= 1)
        rsum += __shfl_xor(rsum, d, 32);
      alpha[r] = __expf(mi[r] - mnew);
      li[r] = li[r] * alpha[r] + rsum;
      mi[r] = mnew;
      plds[(r + 8 * half) * 32 + row]      = (__bf16)p0;  // P -> LDS (D layout)
      plds[(r + 8 * half) * 32 + 16 + row] = (__bf16)p1;
    }

#pragma unroll
    for (int nt = 0; nt < 4; ++nt)
#pragma unroll
      for (int r = 0; r < 8; ++r) o[nt][r] *= alpha[r];

    __syncthreads();                                     // order LDS write->read
    const v16bf pa = frag_from_bf16(plds + row * 32, half);  // P as A-frag
#pragma unroll
    for (int nt = 0; nt < 4; ++nt) {
      const __bf16* vrow = vtb + ((size_t)b * kH + nt * 16 + row) * kT + j0;
      o[nt] = wmma_bf16(pa, frag_from_bf16(vrow, half), o[nt]);
    }
    __syncthreads();                                     // before next overwrite
  }

#pragma unroll
  for (int nt = 0; nt < 4; ++nt) {
#pragma unroll
    for (int r = 0; r < 8; ++r) {
      const int t = t0 + r + 8 * half;
      out[(size_t)(b * kT + t) * kH + nt * 16 + row] = o[nt][r] / li[r];
    }
  }
}

extern "C" void kernel_launch(void* const* d_in, const int* in_sizes, int n_in,
                              void* d_out, int out_size, void* d_ws, size_t ws_size,
                              hipStream_t stream) {
  (void)in_sizes; (void)n_in; (void)out_size; (void)ws_size;
  const float* x  = (const float*)d_in[0];
  const float* Wq = (const float*)d_in[1];
  const float* Wk = (const float*)d_in[2];
  const float* Wv = (const float*)d_in[3];
  float* out = (float*)d_out;

  // workspace: wb[3,H,C] (384KB) + q[B,T,H] + k[B,T,H] + vT[B,H,T] (2MB each)
  __bf16* wbp = (__bf16*)d_ws;
  __bf16* qb  = wbp + (size_t)3 * kH * kC;
  __bf16* kbp = qb  + (size_t)kB * kT * kH;
  __bf16* vtb = kbp + (size_t)kB * kT * kH;

  {
    const int nthreads = (3 * kH * kC) / 4;  // 4 elems per thread
    cvt_w_kernel<<<(nthreads + 255) / 256, 256, 0, stream>>>(Wq, Wk, Wv, wbp);
  }

  qkv_proj_kernel<<<kB * kT / 16, 32, 0, stream>>>(x, wbp, qb, kbp, vtb);

  dim3 gAttn(kT / 16, kB);
  attn_kernel<<<gAttn, 32, 0, stream>>>(qb, kbp, vtb, out);
}
